// enc_rnn3_7868380086825
// MI455X (gfx1250) — matrix operand order repaired and checked by reading, not compiled, not run on hardware
//
#include <hip/hip_runtime.h>

typedef __attribute__((ext_vector_type(16))) _Float16 v16h;
typedef __attribute__((ext_vector_type(8)))  _Float16 v8h;
typedef __attribute__((ext_vector_type(8)))  float    v8f;

#define WAVE (threadIdx.x >> 5)
#define LANE (threadIdx.x & 31)

__device__ __forceinline__ v8f wmma32(v16h a, v16h b, v8f c) {
  // v_wmma_f32_16x16x32_f16: (neg_a, A, neg_b, B, c_mod, C, reuse_a, reuse_b)
  return __builtin_amdgcn_wmma_f32_16x16x32_f16(false, a, false, b, (short)0, c, false, false);
}

// A fragment: 16x32 (MxK) row-major tile, lda in f16 elements; lda % 16 == 0.
// Lane L(0-15): row M=L, K in {0..7, 16..23}; lane L+16: row M=L, K in {8..15, 24..31}.
// Two 16-byte vector loads per lane.
__device__ __forceinline__ v16h load_a(const _Float16* A, int lda) {
  const int lane = LANE, row = lane & 15, hf = lane >> 4;
  const _Float16* p = A + row * lda + (hf << 3);
  v8h lo = *(const v8h*)p;          // K = hf*8 + 0..7
  v8h hi = *(const v8h*)(p + 16);   // K = 16 + hf*8 + 0..7
  return __builtin_shufflevector(lo, hi, 0, 1, 2, 3, 4, 5, 6, 7,
                                         8, 9, 10, 11, 12, 13, 14, 15);
}

// B fragment from [n][k] storage (K contiguous per column), ldk % 16 == 0.
// Lane's fragment at fixed n: K in {8h+0..7, 16+8h+0..7}, h = (lane>=16) (matches load_a).
__device__ __forceinline__ v16h load_bk(const _Float16* Bt, int ldk) {
  const int lane = LANE, n = lane & 15, hf = lane >> 4;
  const _Float16* p = Bt + n * ldk + (hf << 3);
  v8h lo = *(const v8h*)p;
  v8h hi = *(const v8h*)(p + 16);
  return __builtin_shufflevector(lo, hi, 0, 1, 2, 3, 4, 5, 6, 7,
                                         8, 9, 10, 11, 12, 13, 14, 15);
}

// C/D 16x16 f32: VGPR r -> row (8*(lane>=16) + r), col = lane&15.
__device__ __forceinline__ void store_c(float* C, int ldc, v8f c) {
  const int lane = LANE, col = lane & 15, r0 = (lane >> 4) << 3;
#pragma unroll
  for (int r = 0; r < 8; ++r) C[(r0 + r) * ldc + col] = c[r];
}

__device__ __forceinline__ float sigmoidf_(float x) { return 1.0f / (1.0f + __expf(-x)); }

// ---------------------------------------------------------------------------
// Kernel 1/2: time embedding + E x E projection (q or k), f16 output.
// One block = 128 rows. out[row][e] = emb(row) @ W^T + bias, cast to f16.
// W stays in natural row-major [e_out][j] layout == B's [n][k] layout.
// ---------------------------------------------------------------------------
__global__ void __launch_bounds__(256) embed_project_kernel(
    const float* __restrict__ times, const float* __restrict__ W,
    const float* __restrict__ bias, _Float16* __restrict__ out) {
  __shared__ _Float16 sEmb[128 * 144];
  __shared__ _Float16 sW[128 * 144];
  const int r0 = blockIdx.x * 128;
  for (int idx = threadIdx.x; idx < 128 * 128; idx += 256) {
    int row = idx >> 7, e = idx & 127;
    float t = times[r0 + row];
    // div_term = exp(-(2j)*ln(10)/128); even e -> sin, odd -> cos
    float freq = __expf(-(float)(e & ~1) * (2.302585093f / 128.0f));
    float ang = 48.0f * t * freq;
    sEmb[row * 144 + e] = (_Float16)((e & 1) ? __cosf(ang) : __sinf(ang));
    sW[row * 144 + e] = (_Float16)W[idx];  // [e_out][j], coalesced
  }
  __syncthreads();
  const int wv = WAVE, lane = LANE, hf = lane >> 4, nn_ = lane & 15;
#pragma unroll
  for (int i = 0; i < 8; ++i) {
    int tl = wv * 8 + i, mi = tl >> 3, ni = tl & 7;
    v8f acc = {};
#pragma unroll
    for (int ko = 0; ko < 4; ++ko) {
      v16h a = load_a(sEmb + mi * 16 * 144 + ko * 32, 144);
      v16h b = load_bk(sW + ni * 16 * 144 + ko * 32, 144);
      acc = wmma32(a, b, acc);
    }
    int col = ni * 16 + nn_;
    int rbase = mi * 16 + (hf << 3);
    float bcol = bias[col];
#pragma unroll
    for (int r = 0; r < 8; ++r)
      out[(size_t)(r0 + rbase + r) * 128 + col] = (_Float16)(acc[r] + bcol);
  }
}

// ---------------------------------------------------------------------------
// Kernel 3: fused attention.  Per block: one batch b, 64 query rows.
// Streams keys in chunks of 64: S = q k^T / sqrt(E); E = exp(S); N += E @ x.
// Then out = N / N[denom-channel];  h = out @ Wc^T + bc.
// k is read from global in [n][k] form (row-major k!), x staged transposed.
// ---------------------------------------------------------------------------
__global__ void __launch_bounds__(256) attention_kernel(
    const _Float16* __restrict__ qf, const _Float16* __restrict__ kf,
    const float* __restrict__ x, const float* __restrict__ Wc,
    const float* __restrict__ bc, float* __restrict__ h_out) {
  __shared__ _Float16 sQ[64 * 144];     // A: [q][e]
  __shared__ _Float16 sE[64 * 80];      // A: [q][key]
  __shared__ _Float16 sXt[128 * 80];    // B: [c][key]
  __shared__ float    sN[64 * 132];
  __shared__ _Float16 sWc[128 * 144];   // B: [c_out][j]
  __shared__ _Float16 sO[64 * 144];     // A: [q][c]
  const int b = blockIdx.x >> 1;
  const int q0 = (blockIdx.x & 1) * 64;
  for (int idx = threadIdx.x; idx < 64 * 128; idx += 256) {
    int row = idx >> 7, e = idx & 127;
    sQ[row * 144 + e] = qf[(size_t)(q0 + row) * 128 + e];
  }
  for (int idx = threadIdx.x; idx < 128 * 128; idx += 256) {
    int c = idx >> 7, j = idx & 127;
    sWc[c * 144 + j] = (_Float16)Wc[idx];  // coalesced, natural layout
  }
  const int wv = WAVE, lane = LANE, hf = lane >> 4, nn_ = lane & 15;
  v8f accN[4] = {{}, {}, {}, {}};
  const float scale = 0.08838834764f;  // 1/sqrt(128)
  for (int kc = 0; kc < 512; kc += 64) {
    __syncthreads();  // protect sE/sXt of previous chunk (and initial fills)
    const float* xb = x + ((size_t)b * 512 + kc) * 128;
    for (int idx = threadIdx.x; idx < 64 * 128; idx += 256) {
      int kr = idx >> 7, c = idx & 127;
      sXt[c * 80 + kr] = (_Float16)xb[idx];  // transpose into [c][key]
    }
    if (kc + 64 < 512)  // pull next x chunk toward L2/L0 while we do WMMA
      __builtin_prefetch(xb + 64 * 128 + threadIdx.x * 32, 0, 1);
    __syncthreads();
    // S chunk: 64q x 64k, then exp -> sE (f16)
#pragma unroll
    for (int i = 0; i < 2; ++i) {
      int tl = wv * 2 + i, smi = tl >> 2, sni = tl & 3;
      v8f acc = {};
      // B operand straight from global: kf is row-major [key][e] == [n][k]
      const _Float16* kbase = kf + ((size_t)b * 512 + kc + sni * 16) * 128;
#pragma unroll
      for (int ko = 0; ko < 4; ++ko) {
        v16h a = load_a(sQ + smi * 16 * 144 + ko * 32, 144);
        v16h bb = load_bk(kbase + ko * 32, 128);
        acc = wmma32(a, bb, acc);
      }
      int col = sni * 16 + nn_;
      int rbase = smi * 16 + (hf << 3);
#pragma unroll
      for (int r = 0; r < 8; ++r)
        sE[(rbase + r) * 80 + col] = (_Float16)__expf(acc[r] * scale);
    }
    __syncthreads();
    // N += E @ X   (64q x 128c, K = 64 keys)
#pragma unroll
    for (int i = 0; i < 4; ++i) {
      int tl = wv * 4 + i, mi = tl >> 3, ni = tl & 7;
#pragma unroll
      for (int ko = 0; ko < 2; ++ko) {
        v16h a = load_a(sE + mi * 16 * 80 + ko * 32, 80);
        v16h bb = load_bk(sXt + ni * 16 * 80 + ko * 32, 80);
        accN[i] = wmma32(a, bb, accN[i]);
      }
    }
  }
#pragma unroll
  for (int i = 0; i < 4; ++i) {
    int tl = wv * 4 + i, mi = tl >> 3, ni = tl & 7;
    store_c(sN + mi * 16 * 132 + ni * 16, 132, accN[i]);
  }
  __syncthreads();
  // out[q][c] = N[q][c] / N[q][64 + (c & 63)]   (denominator = exp-weighted mask sum)
  for (int idx = threadIdx.x; idx < 64 * 128; idx += 256) {
    int row = idx >> 7, c = idx & 127;
    float num = sN[row * 132 + c];
    float den = sN[row * 132 + 64 + (c & 63)];
    sO[row * 144 + c] = (_Float16)((den > 0.0f) ? (num / den) : 0.0f);
  }
  __syncthreads();
  // h = out @ Wc^T + bc
#pragma unroll
  for (int i = 0; i < 4; ++i) {
    int tl = wv * 4 + i, mi = tl >> 3, ni = tl & 7;
    v8f acc = {};
#pragma unroll
    for (int ko = 0; ko < 4; ++ko) {
      v16h a = load_a(sO + mi * 16 * 144 + ko * 32, 144);
      v16h bb = load_bk(sWc + ni * 16 * 144 + ko * 32, 144);
      acc = wmma32(a, bb, acc);
    }
    int col = ni * 16 + nn_;
    int rbase = mi * 16 + (hf << 3);
    float bcol = bc[col];
#pragma unroll
    for (int r = 0; r < 8; ++r)
      h_out[((size_t)b * 128 + q0 + rbase + r) * 128 + col] = acc[r] + bcol;
  }
}

// ---------------------------------------------------------------------------
// Kernel 4: GI = H @ Wih^T + bih for both directions (parallel part of GRU).
// grid (32 row-blocks of 64, 3 gate-chunks of 128, 2 dirs).
// ---------------------------------------------------------------------------
__global__ void __launch_bounds__(256) gi_kernel(
    const float* __restrict__ h, const float* __restrict__ Wih_f,
    const float* __restrict__ bih_f, const float* __restrict__ Wih_b,
    const float* __restrict__ bih_b, float* __restrict__ gi) {
  __shared__ _Float16 sH[64 * 144];    // A: [row][j]
  __shared__ _Float16 sW[128 * 144];   // B: [c_out local][j]
  const int r0 = blockIdx.x * 64;
  const int c0 = blockIdx.y * 128;
  const int dir = blockIdx.z;
  const float* W = dir ? Wih_b : Wih_f;
  const float* bias = dir ? bih_b : bih_f;
  for (int idx = threadIdx.x; idx < 64 * 128; idx += 256) {
    int row = idx >> 7, j = idx & 127;
    sH[row * 144 + j] = (_Float16)h[(size_t)(r0 + row) * 128 + j];
  }
  for (int idx = threadIdx.x; idx < 128 * 128; idx += 256) {
    int cl = idx >> 7, j = idx & 127;
    sW[cl * 144 + j] = (_Float16)W[(size_t)(c0 + cl) * 128 + j];  // coalesced
  }
  __syncthreads();
  const int wv = WAVE, lane = LANE, hf = lane >> 4, nn_ = lane & 15;
#pragma unroll
  for (int i = 0; i < 4; ++i) {
    int tl = wv * 4 + i, mi = tl >> 3, ni = tl & 7;
    v8f acc = {};
#pragma unroll
    for (int ko = 0; ko < 4; ++ko) {
      v16h a = load_a(sH + mi * 16 * 144 + ko * 32, 144);
      v16h bb = load_bk(sW + ni * 16 * 144 + ko * 32, 144);
      acc = wmma32(a, bb, acc);
    }
    int col = ni * 16 + nn_;
    int rbase = mi * 16 + (hf << 3);
    float bcol = bias[c0 + col];
#pragma unroll
    for (int r = 0; r < 8; ++r)
      gi[((size_t)dir * 2048 + r0 + rbase + r) * 384 + c0 + col] = acc[r] + bcol;
  }
}

// ---------------------------------------------------------------------------
// Kernel 5: sequential GRU, one block per direction. Whh lives in LDS in its
// natural [g][j] row-major layout (== B's [n][k]), ~110 KB (CDNA5 only).
// Each step: gh = h @ Whh^T (WMMA, M=16), fused gates.
// ---------------------------------------------------------------------------
__global__ void __launch_bounds__(256) gru_kernel(
    const float* __restrict__ gi, const float* __restrict__ Whh_f,
    const float* __restrict__ bhh_f, const float* __restrict__ Whh_b,
    const float* __restrict__ bhh_b, float* __restrict__ hh) {
  __shared__ _Float16 sWg[384 * 144];   // B: [g][j]
  __shared__ _Float16 sHh[16 * 144];    // A: h state as f16 (M=16)
  __shared__ float    sHf[16 * 128];    // h state f32
  __shared__ float    sGH[16 * 392];    // gh output
  const int dir = blockIdx.x;
  const float* W = dir ? Whh_b : Whh_f;
  const float* bias = dir ? bhh_b : bhh_f;
  for (int idx = threadIdx.x; idx < 384 * 128; idx += 256) {
    int g = idx >> 7, j = idx & 127;
    sWg[g * 144 + j] = (_Float16)W[idx];  // straight coalesced copy
  }
  for (int idx = threadIdx.x; idx < 16 * 128; idx += 256) sHf[idx] = 0.0f;
  __syncthreads();
  const int wv = WAVE;
  for (int t = 0; t < 128; ++t) {
    const int t_act = dir ? (127 - t) : t;
    for (int idx = threadIdx.x; idx < 16 * 128; idx += 256) {
      int bi = idx >> 7, j = idx & 127;
      sHh[bi * 144 + j] = (_Float16)sHf[idx];
    }
    __syncthreads();
    // gh = h @ Whh^T : M=16, N=384, K=128 -> 24 n-tiles, 3 per wave
#pragma unroll
    for (int i = 0; i < 3; ++i) {
      int ni = wv * 3 + i;
      v8f acc = {};
#pragma unroll
      for (int ko = 0; ko < 4; ++ko) {
        v16h a = load_a(sHh + ko * 32, 144);
        v16h bb = load_bk(sWg + ni * 16 * 144 + ko * 32, 144);
        acc = wmma32(a, bb, acc);
      }
      store_c(sGH + ni * 16, 392, acc);
    }
    __syncthreads();
    // gates
    for (int idx = threadIdx.x; idx < 16 * 128; idx += 256) {
      int bi = idx >> 7, n = idx & 127;
      const float* gib = gi + ((size_t)dir * 2048 + bi * 128 + t_act) * 384;
      float r = sigmoidf_(gib[n] + sGH[bi * 392 + n] + bias[n]);
      float z = sigmoidf_(gib[128 + n] + sGH[bi * 392 + 128 + n] + bias[128 + n]);
      float nv = tanhf(gib[256 + n] + r * (sGH[bi * 392 + 256 + n] + bias[256 + n]));
      float h2 = (1.0f - z) * nv + z * sHf[idx];
      sHf[idx] = h2;
      hh[((size_t)bi * 128 + t_act) * 256 + dir * 128 + n] = h2;
    }
    __syncthreads();
  }
}

// ---------------------------------------------------------------------------
// Kernel 6: z = relu(hh @ W1^T + b1) @ W2^T + b2.  N padded 50->64, 40->48
// with zeroed weight rows; only valid columns stored.
// ---------------------------------------------------------------------------
__global__ void __launch_bounds__(256) mlp_kernel(
    const float* __restrict__ hh, const float* __restrict__ W1,
    const float* __restrict__ b1, const float* __restrict__ W2,
    const float* __restrict__ b2, float* __restrict__ zout) {
  __shared__ _Float16 sHH[64 * 272];   // A: [row][j], K=256
  __shared__ _Float16 sW1[64 * 272];   // B: [n][j] (n: 0..63, rows >=50 zero)
  __shared__ _Float16 sZ1[64 * 80];    // A: [row][n], K=64
  __shared__ _Float16 sW2[48 * 80];    // B: [n][j] (zero-padded)
  const int r0 = blockIdx.x * 64;
  for (int idx = threadIdx.x; idx < 64 * 256; idx += 256) {
    int row = idx >> 8, j = idx & 255;
    sHH[row * 272 + j] = (_Float16)hh[(size_t)(r0 + row) * 256 + j];
  }
  for (int idx = threadIdx.x; idx < 64 * 256; idx += 256) {
    int n = idx >> 8, j = idx & 255;
    sW1[n * 272 + j] = (_Float16)((n < 50) ? W1[n * 256 + j] : 0.0f);
  }
  for (int idx = threadIdx.x; idx < 48 * 64; idx += 256) {
    int n = idx >> 6, j = idx & 63;
    sW2[n * 80 + j] = (_Float16)(((n < 40) && (j < 50)) ? W2[n * 50 + j] : 0.0f);
  }
  __syncthreads();
  const int wv = WAVE, lane = LANE, hf = lane >> 4, nn_ = lane & 15;
  // GEMM1: 64 x 64, K=256
#pragma unroll
  for (int i = 0; i < 2; ++i) {
    int tl = wv * 2 + i, mi = tl >> 2, ni = tl & 3;
    v8f acc = {};
#pragma unroll
    for (int ko = 0; ko < 8; ++ko) {
      v16h a = load_a(sHH + mi * 16 * 272 + ko * 32, 272);
      v16h bb = load_bk(sW1 + ni * 16 * 272 + ko * 32, 272);
      acc = wmma32(a, bb, acc);
    }
    int col = ni * 16 + nn_;
    int rbase = mi * 16 + (hf << 3);
    float bcol = (col < 50) ? b1[col] : 0.0f;
#pragma unroll
    for (int r = 0; r < 8; ++r) {
      float v = acc[r] + bcol;
      v = (col < 50) ? fmaxf(v, 0.0f) : 0.0f;
      sZ1[(rbase + r) * 80 + col] = (_Float16)v;
    }
  }
  __syncthreads();
  // GEMM2: 64 x 48, K=64 (zero-padded); 12 tiles over 8 waves
  for (int tl = wv; tl < 12; tl += 8) {
    int mi = tl / 3, ni = tl % 3;
    v8f acc = {};
#pragma unroll
    for (int ko = 0; ko < 2; ++ko) {
      v16h a = load_a(sZ1 + mi * 16 * 80 + ko * 32, 80);
      v16h bb = load_bk(sW2 + ni * 16 * 80 + ko * 32, 80);
      acc = wmma32(a, bb, acc);
    }
    int col = ni * 16 + nn_;
    int rbase = mi * 16 + (hf << 3);
    if (col < 40) {
      float bcol = b2[col];
#pragma unroll
      for (int r = 0; r < 8; ++r)
        zout[(size_t)(r0 + rbase + r) * 40 + col] = acc[r] + bcol;
    }
  }
}

// ---------------------------------------------------------------------------
extern "C" void kernel_launch(void* const* d_in, const int* in_sizes, int n_in,
                              void* d_out, int out_size, void* d_ws, size_t ws_size,
                              hipStream_t stream) {
  (void)in_sizes; (void)n_in; (void)out_size; (void)ws_size;
  const float* x           = (const float*)d_in[0];
  const float* time_steps  = (const float*)d_in[1];
  const float* query_times = (const float*)d_in[2];
  const float* Wq = (const float*)d_in[3];   const float* bq = (const float*)d_in[4];
  const float* Wk = (const float*)d_in[5];   const float* bk = (const float*)d_in[6];
  const float* Wc = (const float*)d_in[7];   const float* bc = (const float*)d_in[8];
  const float* Wih_f = (const float*)d_in[9];  const float* Whh_f = (const float*)d_in[10];
  const float* bih_f = (const float*)d_in[11]; const float* bhh_f = (const float*)d_in[12];
  const float* Wih_b = (const float*)d_in[13]; const float* Whh_b = (const float*)d_in[14];
  const float* bih_b = (const float*)d_in[15]; const float* bhh_b = (const float*)d_in[16];
  const float* W1 = (const float*)d_in[17];  const float* b1 = (const float*)d_in[18];
  const float* W2 = (const float*)d_in[19];  const float* b2 = (const float*)d_in[20];

  char* ws = (char*)d_ws;
  _Float16* qf    = (_Float16*)(ws + 0);        // 128*128 f16        = 32 KB
  _Float16* kf    = (_Float16*)(ws + 32768);    // 16*512*128 f16     = 2 MB
  float*    h_ws  = (float*)(ws + 2129920);     // 2048*128 f32       = 1 MB
  float*    gi_ws = (float*)(ws + 3178496);     // 2*2048*384 f32     = 6.29 MB
  float*    hh_ws = (float*)(ws + 9469952);     // 2048*256 f32       = 2 MB

  embed_project_kernel<<<1, 256, 0, stream>>>(query_times, Wq, bq, qf);
  embed_project_kernel<<<64, 256, 0, stream>>>(time_steps, Wk, bk, kf);
  attention_kernel<<<32, 256, 0, stream>>>(qf, kf, x, Wc, bc, h_ws);
  gi_kernel<<<dim3(32, 3, 2), 256, 0, stream>>>(h_ws, Wih_f, bih_f, Wih_b, bih_b, gi_ws);
  gru_kernel<<<2, 256, 0, stream>>>(gi_ws, Whh_f, bhh_f, Whh_b, bhh_b, hh_ws);
  mlp_kernel<<<32, 256, 0, stream>>>(hh_ws, W1, b1, W2, b2, (float*)d_out);
}
